// SelfAttention_77111842832741
// MI455X (gfx1250) — matrix operand order repaired and checked by reading, not compiled, not run on hardware
//
#include <hip/hip_runtime.h>

// ---------------------------------------------------------------------------
// CDNA5 (gfx1250, wave32) self-attention with "expressive" normalization.
// Matmuls via v_wmma_f32_16x16x32_bf16 (fp32 accumulation); K-tile staging via
// async global->LDS DMA (global_load_async_to_lds_b128 + s_wait_asynccnt,
// hand-emitted inline asm per the CDNA5 ISA since the builtin is absent);
// alibi stream kept hot with global_prefetch_b8.
// ---------------------------------------------------------------------------

typedef __bf16 bf16;
typedef __attribute__((ext_vector_type(16))) __bf16 v16bf;
typedef __attribute__((ext_vector_type(8)))  __bf16 v8bf;
typedef __attribute__((ext_vector_type(8)))  float  v8f;

#define S_LEN 2048
#define D_DIM 1024
#define H_NUM 16
#define HD    64
#define B_NUM 2

// Async global->LDS path: inline asm, device pass only.
#if defined(__HIP_DEVICE_COMPILE__) && defined(__gfx1250__)
#define USE_ASYNC_LDS 1
#else
#define USE_ASYNC_LDS 0
#endif

union Frag { v16bf v; v8bf h[2]; };

// A-fragment (16x32, bf16): lane l holds row (l&15); element e -> K =
// 8*(l>>4) + (e<8 ? e : e+8). Caller passes p = rowbase + k0 + 8*(lane>>4).
__device__ __forceinline__ v16bf load_a_frag(const bf16* p) {
  Frag f;
  f.h[0] = *reinterpret_cast<const v8bf*>(p);
  f.h[1] = *reinterpret_cast<const v8bf*>(p + 16);
  return f.v;
}

// B-fragment (32x16, bf16), sourced from "rows of B^T" (i.e. W[n][k] rows):
// lane l holds column (l&15); element e -> K = 8*(l>>4) + (e<8 ? e : e+8).
// Caller passes p = colbase + k0 + 8*(lane>>4).
__device__ __forceinline__ v16bf load_b_frag(const bf16* p) {
  Frag f;
  f.h[0] = *reinterpret_cast<const v8bf*>(p);
  f.h[1] = *reinterpret_cast<const v8bf*>(p + 16);
  return f.v;
}

__global__ __launch_bounds__(256)
void f32_to_bf16_kernel(const float* __restrict__ in, bf16* __restrict__ out, int n) {
  int i = blockIdx.x * 256 + threadIdx.x;
  if (i < n) out[i] = (bf16)in[i];
}

// ---------------------------------------------------------------------------
// GEMM: out[m,n] = sum_k A[m,k] * W[n,k] + bias[n]   (y = A @ W^T + b)
// A: MxK bf16, W: NxK bf16. mode 0 -> bf16 out, mode 1 -> fp32 out.
// Block = 256 threads = 8 waves; block tile 128x64; wave tile 32x32.
// ---------------------------------------------------------------------------
__global__ __launch_bounds__(256)
void gemm_bf16_kernel(const bf16* __restrict__ A, const bf16* __restrict__ W,
                      const float* __restrict__ bias,
                      bf16* __restrict__ outb, float* __restrict__ outf,
                      int M, int N, int K, int mode) {
  const int lane = threadIdx.x & 31;
  const int w    = threadIdx.x >> 5;
  const int m0   = blockIdx.x * 128 + (w >> 1) * 32;
  const int n0   = blockIdx.y * 64  + (w & 1) * 32;
  const int rsel = lane & 15;
  const int kbA  = 8  * (lane >> 4);
  const int kbB  = 8  * (lane >> 4);

  const bf16* Ar0 = A + (size_t)(m0 + rsel) * K;
  const bf16* Ar1 = A + (size_t)(m0 + 16 + rsel) * K;
  const bf16* Wr0 = W + (size_t)(n0 + rsel) * K;
  const bf16* Wr1 = W + (size_t)(n0 + 16 + rsel) * K;

  v8f acc[2][2] = {};
  for (int k0 = 0; k0 < K; k0 += 32) {
    v16bf a0 = load_a_frag(Ar0 + k0 + kbA);
    v16bf a1 = load_a_frag(Ar1 + k0 + kbA);
    v16bf b0 = load_b_frag(Wr0 + k0 + kbB);
    v16bf b1 = load_b_frag(Wr1 + k0 + kbB);
    acc[0][0] = __builtin_amdgcn_wmma_f32_16x16x32_bf16(false, a0, false, b0, (short)0, acc[0][0], false, false);
    acc[0][1] = __builtin_amdgcn_wmma_f32_16x16x32_bf16(false, a0, false, b1, (short)0, acc[0][1], false, false);
    acc[1][0] = __builtin_amdgcn_wmma_f32_16x16x32_bf16(false, a1, false, b0, (short)0, acc[1][0], false, false);
    acc[1][1] = __builtin_amdgcn_wmma_f32_16x16x32_bf16(false, a1, false, b1, (short)0, acc[1][1], false, false);
  }

  const int rowoff = 8 * (lane >> 4);
  #pragma unroll
  for (int i = 0; i < 2; i++) {
    #pragma unroll
    for (int j = 0; j < 2; j++) {
      const int col = n0 + j * 16 + rsel;
      const float bv = bias[col];
      #pragma unroll
      for (int r = 0; r < 8; r++) {
        const int row = m0 + i * 16 + r + rowoff;
        const float vv = acc[i][j][r] + bv;
        if (mode == 0) outb[(size_t)row * N + col] = (bf16)vv;
        else           outf[(size_t)row * N + col] = vv;
      }
    }
  }
}

// ---------------------------------------------------------------------------
// Streaming "expressive" attention for one (b, h, 128-q-row) tile.
// Each wave owns 16 q rows; the workgroup marches over keys in chunks of 32.
// K tile (32x64 bf16) staged to LDS by async global->LDS DMA (ASYNCcnt);
// V tile transposed into LDS manually (async copy cannot transpose).
//   s = Q K^T + alibi; w = s^2/(1+s^2);  O += w V;  denom += rowsum(w)
// ---------------------------------------------------------------------------
__global__ __launch_bounds__(256)
void attn_kernel(const bf16* __restrict__ Q, const bf16* __restrict__ Kmat,
                 const bf16* __restrict__ V, const float* __restrict__ alibi,
                 bf16* __restrict__ O) {
  // Single shared block so LDS byte offsets are statically known (base 0):
  //   [0,   2048)  Ks : [key][hd]  current 32x64 K tile (4 KB)
  //   [2048,4096)  Vt : [hd][key]  current 64x32 V tile, transposed (4 KB)
  //   [4096,8192)  Wb : per-wave 16x32 attention-weight repack buffers (8 KB)
  __shared__ __align__(16) bf16 smem[8192];
  bf16* Ks = smem;
  bf16* Vt = smem + 2048;

  const int lane = threadIdx.x & 31;
  const int w    = threadIdx.x >> 5;
  const int h    = blockIdx.y;
  const int b    = blockIdx.z;
  const int q0   = blockIdx.x * 128 + w * 16;
  const int rsel   = lane & 15;
  const int kbA    = 8  * (lane >> 4);
  const int kbB    = 8  * (lane >> 4);
  const int rowoff = 8  * (lane >> 4);
  bf16* Wb = smem + 4096 + w * 512;

  const size_t headoff = (size_t)b * S_LEN * D_DIM + (size_t)h * HD;

  // Q fragments for this wave's 16 rows (kept in VGPRs across the key loop)
  const bf16* qrow = Q + headoff + (size_t)(q0 + rsel) * D_DIM;
  const v16bf aq0 = load_a_frag(qrow + 0  + kbA);
  const v16bf aq1 = load_a_frag(qrow + 32 + kbA);

  // cooperative-load lane mapping: 256 threads x 8 bf16 = one 32x64 tile
  const int lkey = threadIdx.x >> 3;        // 0..31
  const int lhd  = (threadIdx.x & 7) * 8;   // 0,8,...,56
  const bf16* Ksrc = Kmat + headoff + (size_t)lkey * D_DIM + lhd;
  const bf16* Vsrc = V    + headoff + (size_t)lkey * D_DIM + lhd;
  const float* arow = alibi + (size_t)h * S_LEN * S_LEN;

#if USE_ASYNC_LDS
  // Per-lane LDS byte offset of this thread's 16B slice of Ks (smem base = 0).
  const unsigned ks_lds_off = (unsigned)((lkey * 64 + lhd) * 2);
#endif

  v8f accO[4] = {};
  v8f dsum = {};

  for (int kc = 0; kc < S_LEN; kc += 32) {
    __syncthreads();   // previous iteration's LDS reads complete

#if USE_ASYNC_LDS
    {
      // GLOBAL_LOAD_ASYNC_TO_LDS_B128 (GV mode): copies 16 B from global vaddr to LDS at vdst.
      // Tracked by ASYNCcnt; overlaps with the V transpose below.
      const unsigned long long ga =
          (unsigned long long)(size_t)(Ksrc + (size_t)kc * D_DIM);
      asm volatile("global_load_async_to_lds_b128 %0, %1, off"
                   :: "v"(ks_lds_off), "v"(ga)
                   : "memory");
    }
#else
    *reinterpret_cast<v8bf*>(&Ks[lkey * 64 + lhd]) =
        *reinterpret_cast<const v8bf*>(Ksrc + (size_t)kc * D_DIM);
#endif

    // V tile: global -> VGPR -> transposed LDS (overlaps the K-tile DMA)
    const v8bf vv = *reinterpret_cast<const v8bf*>(Vsrc + (size_t)kc * D_DIM);
    #pragma unroll
    for (int e = 0; e < 8; e++) Vt[(lhd + e) * 32 + lkey] = vv[e];

    // Keep the dominant alibi stream (268 MB) in flight for the next chunk:
    // lanes cover all 16 q rows x both 64B halves of the next 32-key span.
    if (kc + 32 < S_LEN)
      __builtin_prefetch(arow + (size_t)(q0 + rsel) * S_LEN + (kc + 32) + 16 * (lane >> 4), 0, 1);

#if USE_ASYNC_LDS
    asm volatile("s_wait_asynccnt 0x0" ::: "memory");
#endif
    __syncthreads();

    // ---- scores for two 16-key tiles; stash w (bf16) in per-wave LDS ----
    #pragma unroll
    for (int t = 0; t < 2; t++) {
      const bf16* kr = &Ks[(t * 16 + rsel) * 64];
      v16bf bk0 = load_b_frag(kr + 0  + kbB);
      v16bf bk1 = load_b_frag(kr + 32 + kbB);
      v8f s = {};
      s = __builtin_amdgcn_wmma_f32_16x16x32_bf16(false, aq0, false, bk0, (short)0, s, false, false);
      s = __builtin_amdgcn_wmma_f32_16x16x32_bf16(false, aq1, false, bk1, (short)0, s, false, false);
      #pragma unroll
      for (int r = 0; r < 8; r++) {
        const int qq = q0 + r + rowoff;
        const int kk = kc + t * 16 + rsel;
        const float sv = s[r] + arow[(size_t)qq * S_LEN + kk];
        const float a2 = sv * sv;
        const float wg = a2 / (1.0f + a2);
        dsum[r] += wg;
        Wb[(r + rowoff) * 32 + t * 16 + rsel] = (bf16)wg;
      }
    }

    // ---- O += w @ V  (repacked A-frag from per-wave LDS; same-wave LDS is
    //      in-order, no barrier needed) ----
    const v16bf wa = load_a_frag(&Wb[rsel * 32] + kbA);
    #pragma unroll
    for (int n = 0; n < 4; n++) {
      v16bf bv = load_b_frag(&Vt[(n * 16 + rsel) * 32] + kbB);
      accO[n] = __builtin_amdgcn_wmma_f32_16x16x32_bf16(false, wa, false, bv, (short)0, accO[n], false, false);
    }
  }

  // ---- per-row denominators: reduce across the 16-lane half of the wave ----
  float rinv[8];
  #pragma unroll
  for (int r = 0; r < 8; r++) {
    float d = dsum[r];
    #pragma unroll
    for (int m = 1; m < 16; m <<= 1) d += __shfl_xor(d, m, 32);
    rinv[r] = 1.0f / d;
  }

  #pragma unroll
  for (int n = 0; n < 4; n++) {
    #pragma unroll
    for (int r = 0; r < 8; r++) {
      const int qq = q0 + r + rowoff;
      const int hd = n * 16 + rsel;
      O[headoff + (size_t)qq * D_DIM + hd] = (bf16)(accO[n][r] * rinv[r]);
    }
  }
}

// ---------------------------------------------------------------------------
// Host-side launcher
// ---------------------------------------------------------------------------
extern "C" void kernel_launch(void* const* d_in, const int* in_sizes, int n_in,
                              void* d_out, int out_size, void* d_ws, size_t ws_size,
                              hipStream_t stream) {
  const float* x     = (const float*)d_in[0];
  const float* alibi = (const float*)d_in[1];
  const float* Wq    = (const float*)d_in[2];
  const float* bq    = (const float*)d_in[3];
  const float* Wk    = (const float*)d_in[4];
  const float* bk    = (const float*)d_in[5];
  const float* Wv    = (const float*)d_in[6];
  const float* bv    = (const float*)d_in[7];
  const float* Wo    = (const float*)d_in[8];
  const float* bo    = (const float*)d_in[9];
  float* out = (float*)d_out;

  const size_t MD = (size_t)B_NUM * S_LEN * D_DIM;   // 4 Mi activations
  const size_t WW = (size_t)D_DIM * D_DIM;           // 1 Mi per weight

  char* p = (char*)d_ws;
  bf16* xb  = (bf16*)p; p += MD * 2;
  bf16* wqb = (bf16*)p; p += WW * 2;
  bf16* wkb = (bf16*)p; p += WW * 2;
  bf16* wvb = (bf16*)p; p += WW * 2;
  bf16* wob = (bf16*)p; p += WW * 2;
  bf16* qb  = (bf16*)p; p += MD * 2;
  bf16* kb2 = (bf16*)p; p += MD * 2;
  bf16* vb2 = (bf16*)p; p += MD * 2;
  bf16* ab  = (bf16*)p; p += MD * 2;   // total ~48 MB

  // 1) fp32 -> bf16 conversions
  f32_to_bf16_kernel<<<(int)((MD + 255) / 256), 256, 0, stream>>>(x,  xb,  (int)MD);
  f32_to_bf16_kernel<<<(int)((WW + 255) / 256), 256, 0, stream>>>(Wq, wqb, (int)WW);
  f32_to_bf16_kernel<<<(int)((WW + 255) / 256), 256, 0, stream>>>(Wk, wkb, (int)WW);
  f32_to_bf16_kernel<<<(int)((WW + 255) / 256), 256, 0, stream>>>(Wv, wvb, (int)WW);
  f32_to_bf16_kernel<<<(int)((WW + 255) / 256), 256, 0, stream>>>(Wo, wob, (int)WW);

  // 2) QKV projections: [4096,1024] @ [1024,1024]^T + bias -> bf16
  const dim3 gg(32, 16), blk(256);
  gemm_bf16_kernel<<<gg, blk, 0, stream>>>(xb, wqb, bq, qb,  nullptr, 4096, 1024, 1024, 0);
  gemm_bf16_kernel<<<gg, blk, 0, stream>>>(xb, wkb, bk, kb2, nullptr, 4096, 1024, 1024, 0);
  gemm_bf16_kernel<<<gg, blk, 0, stream>>>(xb, wvb, bv, vb2, nullptr, 4096, 1024, 1024, 0);

  // 3) streaming expressive attention (async-LDS staged K tiles)
  attn_kernel<<<dim3(16, H_NUM, B_NUM), blk, 0, stream>>>(qb, kb2, vb2, alibi, ab);

  // 4) output projection -> fp32 d_out
  gemm_bf16_kernel<<<gg, blk, 0, stream>>>(ab, wob, bo, nullptr, out, 4096, 1024, 1024, 1);
}
